// Embeddings_traj_66271345377352
// MI455X (gfx1250) — matrix operand order repaired and checked by reading, not compiled, not run on hardware
//
#include <hip/hip_runtime.h>
#include <hip/hip_bf16.h>

#define LOCV 10000
#define DLOC 128
#define DSUM 256
#define NTOK (64 * 128)   // B*S
#define NS   128
#define KSTEP 64

typedef __attribute__((ext_vector_type(16))) __bf16 v16bf;
typedef __attribute__((ext_vector_type(8)))  __bf16 v8bf;
typedef __attribute__((ext_vector_type(8)))  float  v8f;

// ---------------------------------------------------------------------------
// Kernel 0: convert W_loc [10000,128] f32 -> Wt [128][10000] bf16 (transposed)
// ---------------------------------------------------------------------------
__global__ __launch_bounds__(256) void wt_convert_kernel(
    const float* __restrict__ W, __bf16* __restrict__ Wt) {
  int idx = blockIdx.x * 256 + threadIdx.x;
  if (idx >= LOCV * DLOC) return;
  int n = idx / LOCV;          // output column (0..127)
  int k = idx % LOCV;          // K row
  Wt[idx] = (__bf16)W[(size_t)k * DLOC + n];
}

// ---------------------------------------------------------------------------
// Kernel 1: M = relu(A @ W + b) * 16   (M: [10000,128] f32 in d_ws)
// Tile: 64(M) x 128(N) per workgroup, 8 waves; each wave: 16M x 64N.
// K loop in steps of 64 (2 bf16-WMMA k-chunks), software-pipelined:
// next tile's global loads are issued before the current tile's WMMAs.
// ---------------------------------------------------------------------------
__global__ __launch_bounds__(256) void gcn_gemm_kernel(
    const float* __restrict__ A, const __bf16* __restrict__ Wt,
    const float* __restrict__ bias, float* __restrict__ Mout) {
  __shared__ __align__(16) __bf16 Asl[64 * KSTEP];    // [64 rows][64 K]
  __shared__ __align__(16) __bf16 Bsl[128 * KSTEP];   // [128 cols][64 K]

  const int tid  = threadIdx.x;
  const int lane = tid & 31;
  const int wave = tid >> 5;
  const int m_base = blockIdx.x * 64;

  const int msub  = wave & 3;           // M subtile (16 rows each)
  const int jbase = (wave >> 2) * 4;    // first of 4 N subtiles (0 or 4)

  v8f acc[4] = {};

  // A staging: 64 rows x 4 chunks-of-16; clamp row (rows >= LOCV feed only
  // never-stored outputs, so garbage there is harmless -> no guard).
  const int arow = tid >> 2;
  const int akq  = tid & 3;
  const int agr  = min(m_base + arow, LOCV - 1);
  const float* __restrict__ Arow = A + (size_t)agr * LOCV;
  // B staging: 128 n-rows x 2 halves-of-32 (Wt already bf16 + transposed)
  const int bn  = tid >> 1;
  const int bkh = tid & 1;
  const __bf16* __restrict__ Wrow = Wt + (size_t)bn * LOCV;

  const int half = lane >> 4;           // 0: lanes 0-15, 1: lanes 16-31
  const int l16  = lane & 15;
  const __bf16* __restrict__ abase = &Asl[(msub * 16 + l16) * KSTEP];

  // staging registers (pipeline stage)
  float4 ar0, ar1, ar2, ar3;
  v8bf   br0, br1, br2, br3;

  auto load_tile = [&](int k0) {
    const float4* pa = (const float4*)(Arow + k0 + akq * 16);
    ar0 = pa[0]; ar1 = pa[1]; ar2 = pa[2]; ar3 = pa[3];
    const v8bf* pb = (const v8bf*)(Wrow + k0 + bkh * 32);
    br0 = pb[0]; br1 = pb[1]; br2 = pb[2]; br3 = pb[3];
  };

  auto store_tile = [&]() {
    __bf16 tmp[16];
    tmp[0]  = (__bf16)ar0.x; tmp[1]  = (__bf16)ar0.y;
    tmp[2]  = (__bf16)ar0.z; tmp[3]  = (__bf16)ar0.w;
    tmp[4]  = (__bf16)ar1.x; tmp[5]  = (__bf16)ar1.y;
    tmp[6]  = (__bf16)ar1.z; tmp[7]  = (__bf16)ar1.w;
    tmp[8]  = (__bf16)ar2.x; tmp[9]  = (__bf16)ar2.y;
    tmp[10] = (__bf16)ar2.z; tmp[11] = (__bf16)ar2.w;
    tmp[12] = (__bf16)ar3.x; tmp[13] = (__bf16)ar3.y;
    tmp[14] = (__bf16)ar3.z; tmp[15] = (__bf16)ar3.w;
    v8bf* qa = (v8bf*)(&Asl[arow * KSTEP + akq * 16]);
    qa[0] = ((const v8bf*)tmp)[0];
    qa[1] = ((const v8bf*)tmp)[1];
    v8bf* qb = (v8bf*)(&Bsl[bn * KSTEP + bkh * 32]);
    qb[0] = br0; qb[1] = br1; qb[2] = br2; qb[3] = br3;
  };

  auto compute = [&]() {
#pragma unroll
    for (int c = 0; c < 2; ++c) {
      // A frag (contiguous K order, matches B): elems 0-15: K=c*32+half*16+{0..15}
      const __bf16* ab = abase + c * 32;
      v8bf alo = *(const v8bf*)(ab + half * 16);
      v8bf ahi = *(const v8bf*)(ab + half * 16 + 8);
      v16bf afrag = __builtin_shufflevector(alo, ahi,
          0, 1, 2, 3, 4, 5, 6, 7, 8, 9, 10, 11, 12, 13, 14, 15);
      // load all 4 B frags first so the 4 WMMAs can issue back-to-back
      v16bf bfrag[4];
#pragma unroll
      for (int j = 0; j < 4; ++j) {
        const __bf16* bb =
            &Bsl[((jbase + j) * 16 + l16) * KSTEP + c * 32 + half * 16];
        v8bf blo = *(const v8bf*)(bb);
        v8bf bhi = *(const v8bf*)(bb + 8);
        bfrag[j] = __builtin_shufflevector(blo, bhi,
            0, 1, 2, 3, 4, 5, 6, 7, 8, 9, 10, 11, 12, 13, 14, 15);
      }
#pragma unroll
      for (int j = 0; j < 4; ++j) {
        acc[j] = __builtin_amdgcn_wmma_f32_16x16x32_bf16(
            false, afrag, false, bfrag[j], (short)0, acc[j], false, false);
      }
    }
  };

  const int nFull = LOCV / KSTEP;       // 156 unguarded iterations

  load_tile(0);                         // prologue
  for (int kt = 0; kt < nFull; ++kt) {
    store_tile();                       // consume pipeline regs -> LDS
    __syncthreads();
    if (kt + 1 < nFull) load_tile((kt + 1) * KSTEP);  // prefetch next tile
    compute();                          // WMMAs overlap in-flight globals
    __syncthreads();
  }

  // ---- guarded K tail: K = 9984..9999 valid, rest zero-filled (adds 0)
  {
    const int k0 = nFull * KSTEP;
    {
      __bf16 tmp[16];
      for (int i = 0; i < 16; ++i) {
        const int kk = k0 + akq * 16 + i;
        tmp[i] = (kk < LOCV) ? (__bf16)Arow[kk] : (__bf16)0.f;
      }
      v8bf* qa = (v8bf*)(&Asl[arow * KSTEP + akq * 16]);
      qa[0] = ((const v8bf*)tmp)[0];
      qa[1] = ((const v8bf*)tmp)[1];
    }
    {
      __bf16 tmp[32];
      for (int i = 0; i < 32; ++i) {
        const int kk = k0 + bkh * 32 + i;
        tmp[i] = (kk < LOCV) ? Wrow[kk] : (__bf16)0.f;
      }
      v8bf* qb = (v8bf*)(&Bsl[bn * KSTEP + bkh * 32]);
      qb[0] = ((const v8bf*)tmp)[0];
      qb[1] = ((const v8bf*)tmp)[1];
      qb[2] = ((const v8bf*)tmp)[2];
      qb[3] = ((const v8bf*)tmp)[3];
    }
    __syncthreads();
    compute();
  }

  // ---- epilogue: bias + relu + scale(=sqrt(256)=16), store M
  const int rowb = m_base + msub * 16 + half * 8;   // acc elem r -> row rowb+r
  if (m_base + 64 <= LOCV) {
    // full block: no row guard
#pragma unroll
    for (int j = 0; j < 4; ++j) {
      const int col = (jbase + j) * 16 + l16;
      const float bv = bias[col];
#pragma unroll
      for (int r = 0; r < 8; ++r) {
        float x = acc[j][r] + bv;
        x = x > 0.f ? x : 0.f;
        Mout[(size_t)(rowb + r) * DLOC + col] = x * 16.0f;
      }
    }
  } else {
#pragma unroll
    for (int j = 0; j < 4; ++j) {
      const int col = (jbase + j) * 16 + l16;
      const float bv = bias[col];
#pragma unroll
      for (int r = 0; r < 8; ++r) {
        const int row = rowb + r;
        if (row < LOCV) {
          float x = acc[j][r] + bv;
          x = x > 0.f ? x : 0.f;
          Mout[(size_t)row * DLOC + col] = x * 16.0f;
        }
      }
    }
  }
}

// ---------------------------------------------------------------------------
// Kernel 2: assemble outputs.
//   out_res[token][0:128]   = M[loc[token]]            (relu+scale folded in)
//   out_res[token][128:192] = emb_st[st[token]] * 16
//   out_res[token][192:256] = emb_ed[ed[token]] * 16
//   out_yt [token][0:64]    = emb_st[yt[token]]        (unscaled)
// ---------------------------------------------------------------------------
__global__ __launch_bounds__(256) void gather_kernel(
    const int* __restrict__ loc, const int* __restrict__ st,
    const int* __restrict__ ed, const float* __restrict__ M,
    const float* __restrict__ emb_st, const float* __restrict__ emb_ed,
    float* __restrict__ out_res, float* __restrict__ out_yt) {
  const int token = blockIdx.x;         // 0..8191
  const int t = threadIdx.x;            // 0..255
  const int s = token & (NS - 1);

  float* orow = out_res + (size_t)token * DSUM;
  if (t < 128) {
    orow[t] = M[(size_t)loc[token] * DLOC + t];
  } else if (t < 192) {
    orow[t] = emb_st[st[token] * 64 + (t - 128)] * 16.0f;
  } else {
    orow[t] = emb_ed[ed[token] * 64 + (t - 192)] * 16.0f;
  }
  if (t < 64) {
    const int yt = (s < NS - 1) ? st[token + 1] : 0;
    out_yt[(size_t)token * 64 + t] = emb_st[yt * 64 + t];
  }
}

// ---------------------------------------------------------------------------
extern "C" void kernel_launch(void* const* d_in, const int* in_sizes, int n_in,
                              void* d_out, int out_size, void* d_ws, size_t ws_size,
                              hipStream_t stream) {
  const int*   loc    = (const int*)d_in[0];
  const int*   st     = (const int*)d_in[1];
  const int*   ed     = (const int*)d_in[2];
  const float* A      = (const float*)d_in[3];
  const float* W      = (const float*)d_in[4];
  const float* b      = (const float*)d_in[5];
  const float* emb_st = (const float*)d_in[6];
  const float* emb_ed = (const float*)d_in[7];

  // workspace layout: Mout f32 [10000*128] | Wt bf16 [128*10000]
  float*  Mout = (float*)d_ws;
  __bf16* Wt   = (__bf16*)((char*)d_ws + (size_t)LOCV * DLOC * sizeof(float));

  float* out_res = (float*)d_out;
  float* out_yt  = out_res + (size_t)NTOK * DSUM;

  wt_convert_kernel<<<dim3((LOCV * DLOC + 255) / 256), dim3(256), 0, stream>>>(W, Wt);
  gcn_gemm_kernel<<<dim3((LOCV + 63) / 64), dim3(256), 0, stream>>>(A, Wt, b, Mout);
  gather_kernel<<<dim3(NTOK), dim3(256), 0, stream>>>(loc, st, ed, Mout,
                                                      emb_st, emb_ed, out_res, out_yt);
}
